// Model_40827959116313
// MI455X (gfx1250) — matrix operand order repaired and checked by reading, not compiled, not run on hardware
//
#include <hip/hip_runtime.h>

// ---------------------------------------------------------------------------
// Problem constants (from reference)
// ---------------------------------------------------------------------------
#define Bq   32
#define Nn   128
#define Ee   2048
#define Gg   1024
#define Vv   10000
#define D_IN 20
#define HID  1024
#define MID  128
#define EPSf 1e-5f
#define CAP  64            // per-node neighbor-list capacity (mean deg = 16)

#define LDSROW 136         // 128 + 8 bf16 pad -> 272B row stride (4-bank rotate)
#define KCHUNK 128         // K elements staged per LDS chunk
#define NCHUNK (HID / KCHUNK)

typedef __attribute__((ext_vector_type(16))) __bf16 v16bf;
typedef __attribute__((ext_vector_type(8)))  float  v8f;

// round-to-nearest-even f32 -> bf16 (stored as ushort)
__device__ __forceinline__ unsigned short f2bf(float f) {
  unsigned int u = __float_as_uint(f);
  unsigned int lsb = (u >> 16) & 1u;
  u += 0x7fffu + lsb;
  return (unsigned short)(u >> 16);
}

// ---------------------------------------------------------------------------
// 0) zero the degree counters
// ---------------------------------------------------------------------------
__global__ void k_zero_cnt(int* cnt, int n) {
  int i = blockIdx.x * blockDim.x + threadIdx.x;
  if (i < n) cnt[i] = 0;
}

// 1) build per-node incoming-neighbor lists (shared by both GIN layers)
__global__ void k_build_adj(const int* __restrict__ src, const int* __restrict__ dst,
                            int* __restrict__ adj, int* __restrict__ cnt) {
  int i = blockIdx.x * blockDim.x + threadIdx.x;     // over B*E
  if (i >= Bq * Ee) return;
  int b = i / Ee;
  int s = src[i], d = dst[i];
  int node = b * Nn + d;
  int pos = atomicAdd(&cnt[node], 1);
  if (pos < CAP) adj[node * CAP + pos] = s;
}

// 2) h2 = h @ W_init + b_init   [B*N,20]x[20,1024]
__global__ void k_mlp_init(const float* __restrict__ h, const float* __restrict__ W,
                           const float* __restrict__ bias, float* __restrict__ out) {
  int i = blockIdx.x * blockDim.x + threadIdx.x;     // B*N*HID
  if (i >= Bq * Nn * HID) return;
  int row = i >> 10, f = i & (HID - 1);
  const float* hr = h + row * D_IN;
  float s = bias[f];
#pragma unroll
  for (int k = 0; k < D_IN; ++k) s += hr[k] * W[k * HID + f];
  out[i] = s;
}

// 3) GIN mean: out = in + mean_{in-neighbors}(in)   (one block per node, 128 thr)
__global__ void k_gin_mean(const float* __restrict__ in, float* __restrict__ out,
                           const int* __restrict__ adj, const int* __restrict__ cnt) {
  int node = blockIdx.x;                              // b*Nn + n
  int b = node >> 7;
  int t = threadIdx.x;                                // 128 threads, 8 feats each
  int degr = cnt[node];
  int deg = degr < CAP ? degr : CAP;
  float acc[8] = {0, 0, 0, 0, 0, 0, 0, 0};
  const int* lst = adj + node * CAP;
  for (int j = 0; j < deg; ++j) {
    const float* r = in + ((b << 7) + lst[j]) * HID;
#pragma unroll
    for (int q = 0; q < 8; ++q) acc[q] += r[t + (q << 7)];
  }
  float inv = 1.0f / (float)(deg > 0 ? deg : 1);
  const float* selfr = in + node * HID;
  float* o = out + node * HID;
#pragma unroll
  for (int q = 0; q < 8; ++q) {
    int f = t + (q << 7);
    o[f] = selfr[f] + acc[q] * inv;
  }
}

// 4) BatchNorm over nodes (axis=1) + ReLU, in place. one thread per (b,f)
__global__ void k_bn_nodes_relu(float* __restrict__ x, const float* __restrict__ gamma,
                                const float* __restrict__ beta) {
  int i = blockIdx.x * blockDim.x + threadIdx.x;      // B*HID
  if (i >= Bq * HID) return;
  int b = i >> 10, f = i & (HID - 1);
  float* p = x + b * Nn * HID + f;
  float s = 0.f, sq = 0.f;
  for (int n = 0; n < Nn; ++n) { float v = p[n * HID]; s += v; sq += v * v; }
  float mu = s * (1.0f / Nn);
  float var = sq * (1.0f / Nn) - mu * mu;
  float inv = rsqrtf(var + EPSf);
  float g = gamma[f], be = beta[f];
  for (int n = 0; n < Nn; ++n) {
    float y = g * (p[n * HID] - mu) * inv + be;
    p[n * HID] = fmaxf(y, 0.f);
  }
}

// 5) qemb = mean over nodes
__global__ void k_qemb(const float* __restrict__ x, float* __restrict__ qemb) {
  int i = blockIdx.x * blockDim.x + threadIdx.x;      // B*HID
  if (i >= Bq * HID) return;
  int b = i >> 10, f = i & (HID - 1);
  const float* p = x + b * Nn * HID + f;
  float s = 0.f;
  for (int n = 0; n < Nn; ++n) s += p[n * HID];
  qemb[i] = s * (1.0f / Nn);
}

// 6) qw[b,m] = b_fc[m] + qemb[b,:] @ W_fc[0:HID, m]   (broadcast half of the GEMM)
__global__ void k_qw(const float* __restrict__ qemb, const float* __restrict__ Wfc,
                     const float* __restrict__ bfc, float* __restrict__ qw) {
  int i = blockIdx.x * blockDim.x + threadIdx.x;      // B*MID
  if (i >= Bq * MID) return;
  int b = i >> 7, m = i & (MID - 1);
  float s = bfc[m];
  const float* q = qemb + b * HID;
  for (int k = 0; k < HID; ++k) s += q[k] * Wfc[k * MID + m];
  qw[i] = s;
}

// 7a) allDBGEmb f32 -> bf16
__global__ void k_conv_emb(const float* __restrict__ src, unsigned short* __restrict__ dst, int n) {
  int i = blockIdx.x * blockDim.x + threadIdx.x;
  if (i < n) dst[i] = f2bf(src[i]);
}

// 7b) W_g = W_fc[HID:2*HID, :] -> bf16, transposed to [MID][HID]
__global__ void k_conv_wt(const float* __restrict__ Wfc, unsigned short* __restrict__ Wt) {
  int i = blockIdx.x * blockDim.x + threadIdx.x;      // MID*HID
  if (i >= MID * HID) return;
  int n = i >> 10, k = i & (HID - 1);
  Wt[i] = f2bf(Wfc[(HID + k) * MID + n]);
}

// ---------------------------------------------------------------------------
// async global->LDS staging of one Wt K-chunk (32KB) into a padded LDS buffer.
// Per thread: 8 x b128 transfers, tracked on ASYNCcnt (no VGPR round-trip).
// ---------------------------------------------------------------------------
__device__ __forceinline__ void issue_chunk_async(const unsigned short* __restrict__ Wt,
                                                  int tid, unsigned ldsBufByte, int kc) {
  const unsigned kbyte = (unsigned)(kc * KCHUNK) * 2u;
#pragma unroll
  for (int it = 0; it < 8; ++it) {
    int u  = tid + it * 256;                 // uint4 tile index, 2048 per chunk
    int n  = u >> 4;
    int j8 = (u & 15) << 3;                  // element offset within chunk row
    unsigned goff  = (unsigned)n * (HID * 2u) + kbyte + (unsigned)j8 * 2u;
    unsigned laddr = ldsBufByte + ((unsigned)n * LDSROW + (unsigned)j8) * 2u;
    asm volatile("global_load_async_to_lds_b128 %0, %1, %2"
                 :: "v"(laddr), "v"(goff), "s"(Wt)
                 : "memory");
  }
}

// ---------------------------------------------------------------------------
// 8) Big GEMM: Hpre[m, :] = emb_bf16[gPos[m], :] @ W_g + qw[b(m), :]
//    M = B*G = 32768, K = 1024, N = 128.  bf16 WMMA, fp32 accumulate.
//    wave tile: 16(M) x 128(N); block = 8 waves.
//    Wt K-chunks double-buffered in LDS via async global->LDS (ASYNCcnt),
//    +8-element row pad (272B stride -> 4-bank rotation, conflict free).
// ---------------------------------------------------------------------------
__global__ __launch_bounds__(256) void k_gemm_score(
    const unsigned short* __restrict__ emb,   // [V][HID] bf16
    const unsigned short* __restrict__ Wt,    // [MID][HID] bf16
    const int* __restrict__ gPos,             // [B*G]
    const float* __restrict__ qw,             // [B][MID]
    float* __restrict__ Hpre)                 // [B*G][MID]
{
  __shared__ __attribute__((aligned(16))) unsigned short lds[2 * MID * LDSROW];

  const int tid    = threadIdx.x;
  const int wave   = tid >> 5;
  const int lane   = tid & 31;
  const int lane16 = lane & 15;
  const int hiHalf = lane >> 4;               // 0: lanes 0-15, 1: lanes 16-31
  const int m0     = (blockIdx.x * 8 + wave) * 16;
  const int b      = m0 >> 10;                // stripes never cross a batch row

  // LDS byte offset of the shared buffer (generic LDS ptr: low 32 bits = offset)
  const unsigned ldsBase = (unsigned)(uintptr_t)(&lds[0]);

  // A-matrix row pointer for this lane (row M = lane % 16 of the stripe)
  const long vidx = (long)gPos[m0 + lane16];
  const unsigned short* arow = emb + vidx * HID;

  const v8f vzero = {0.f, 0.f, 0.f, 0.f, 0.f, 0.f, 0.f, 0.f};
  v8f acc[8];
#pragma unroll
  for (int t = 0; t < 8; ++t) acc[t] = vzero;

  // prime buffer 0
  issue_chunk_async(Wt, tid, ldsBase, 0);
  asm volatile("s_wait_asynccnt 0x0" ::: "memory");
  __syncthreads();

  for (int kc = 0; kc < NCHUNK; ++kc) {
    const int buf = kc & 1;
    // prefetch next chunk into the other buffer while we compute on this one;
    // the barrier that ended iteration kc-1 guarantees that buffer is idle.
    if (kc + 1 < NCHUNK)
      issue_chunk_async(Wt, tid, ldsBase + (unsigned)((buf ^ 1) * MID * LDSROW * 2), kc + 1);

    const unsigned short* bbuf = &lds[buf * MID * LDSROW];
    const int kbase = kc * KCHUNK;

#pragma unroll
    for (int kk = 0; kk < 4; ++kk) {          // 4 x K=32 WMMA steps per chunk
      // A fragment: VGPR0-3 hold K=base..base+7 pairs, VGPR4-7 hold base+16..+23
      const int kA = kbase + kk * 32 + (hiHalf ? 8 : 0);
      v16bf a;
      ((uint4*)&a)[0] = *(const uint4*)(arow + kA);
      ((uint4*)&a)[1] = *(const uint4*)(arow + kA + 16);

      // B fragments: lanes 0-15 hold K=k0..k0+7,k0+16..+23 (col = lane), lanes 16-31 K=+8.
      // Distinct fragment registers -> all 16 ds_loads issue before the WMMA
      // chain; waits become partial instead of dscnt==0 per tile.
      const int kB = kk * 32 + (hiHalf ? 8 : 0);
      v16bf bfr[8];
#pragma unroll
      for (int t = 0; t < 8; ++t) {
        const unsigned short* bp = &bbuf[(t * 16 + lane16) * LDSROW + kB];
        ((uint4*)&bfr[t])[0] = *(const uint4*)(bp);
        ((uint4*)&bfr[t])[1] = *(const uint4*)(bp + 16);
      }
#pragma unroll
      for (int t = 0; t < 8; ++t)
        acc[t] = __builtin_amdgcn_wmma_f32_16x16x32_bf16(
            false, a, false, bfr[t], (short)0, acc[t], false, false);
    }

    // next chunk's async transfers must have landed before anyone reads them
    asm volatile("s_wait_asynccnt 0x0" ::: "memory");
    __syncthreads();
  }

  // epilogue: C layout -> VGPR j holds M=j (lanes 0-15) / M=j+8 (lanes 16-31)
#pragma unroll
  for (int t = 0; t < 8; ++t) {
    const int n  = t * 16 + lane16;
    const float qv = qw[b * MID + n];
#pragma unroll
    for (int j = 0; j < 8; ++j) {
      const int m = m0 + j + (hiHalf ? 8 : 0);
      Hpre[(long)m * MID + n] = acc[t][j] + qv;
    }
  }
}

// 9) BN-over-G statistics per (b, mid)
__global__ void k_bn_g_stats(const float* __restrict__ Hpre, float* __restrict__ mu,
                             float* __restrict__ inv) {
  int i = blockIdx.x * blockDim.x + threadIdx.x;      // B*MID
  if (i >= Bq * MID) return;
  int b = i >> 7, m = i & (MID - 1);
  const float* p = Hpre + (long)b * Gg * MID + m;
  float s = 0.f, sq = 0.f;
  for (int g = 0; g < Gg; ++g) { float v = p[g * MID]; s += v; sq += v * v; }
  float muv = s * (1.0f / Gg);
  float var = sq * (1.0f / Gg) - muv * muv;
  mu[i]  = muv;
  inv[i] = rsqrtf(var + EPSf);
}

// 10) head: probs = sigmoid(relu(bn(Hpre)) @ W_fc2 + b_fc2). one block per (b,g)
__global__ void k_head(const float* __restrict__ Hpre, const float* __restrict__ mu,
                       const float* __restrict__ inv, const float* __restrict__ gamma,
                       const float* __restrict__ beta, const float* __restrict__ Wfc2,
                       const float* __restrict__ bfc2, float* __restrict__ out) {
  __shared__ float red[MID];
  int bg = blockIdx.x;                                // b*G + g
  int m  = threadIdx.x;                               // 128 threads
  int bm = (bg >> 10) * MID + m;
  float x = Hpre[(long)bg * MID + m];
  float y = gamma[m] * (x - mu[bm]) * inv[bm] + beta[m];
  y = fmaxf(y, 0.f);
  red[m] = y * Wfc2[m];
  __syncthreads();
#pragma unroll
  for (int off = 64; off > 0; off >>= 1) {
    if (m < off) red[m] += red[m + off];
    __syncthreads();
  }
  if (m == 0) {
    float z = red[0] + bfc2[0];
    out[bg] = 1.0f / (1.0f + __expf(-z));
  }
}

// ---------------------------------------------------------------------------
// launch
// ---------------------------------------------------------------------------
extern "C" void kernel_launch(void* const* d_in, const int* in_sizes, int n_in,
                              void* d_out, int out_size, void* d_ws, size_t ws_size,
                              hipStream_t stream) {
  const float* h        = (const float*)d_in[0];
  const int*   esrc     = (const int*)d_in[1];
  const int*   edst     = (const int*)d_in[2];
  const int*   gPos     = (const int*)d_in[3];
  const float* allEmb   = (const float*)d_in[4];
  const float* W_init   = (const float*)d_in[5];
  const float* b_init   = (const float*)d_in[6];
  const float* gamma1   = (const float*)d_in[7];
  const float* beta1    = (const float*)d_in[8];
  const float* gamma2   = (const float*)d_in[9];
  const float* beta2    = (const float*)d_in[10];
  const float* W_fc     = (const float*)d_in[11];
  const float* b_fc     = (const float*)d_in[12];
  const float* gamma_bn = (const float*)d_in[13];
  const float* beta_bn  = (const float*)d_in[14];
  const float* W_fc2    = (const float*)d_in[15];
  const float* b_fc2    = (const float*)d_in[16];
  float* probs = (float*)d_out;

  // workspace layout (bytes, all 256-aligned)
  char* ws = (char*)d_ws;
  unsigned short* embbf = (unsigned short*)(ws);                    // 20,480,000
  unsigned short* Wt    = (unsigned short*)(ws + 20480000);         //    262,144
  float* hA   = (float*)(ws + 20742144);                            // 16,777,216
  float* hB   = (float*)(ws + 37519360);                            // 16,777,216 (also Hpre)
  int*   adj  = (int*)  (ws + 54296576);                            //  1,048,576
  int*   cnt  = (int*)  (ws + 55345152);                            //     16,384
  float* qemb = (float*)(ws + 55361536);                            //    131,072
  float* qw   = (float*)(ws + 55492608);                            //     16,384
  float* mu   = (float*)(ws + 55508992);                            //     16,384
  float* inv  = (float*)(ws + 55525376);                            //     16,384
  float* Hpre = hB;   // hB lifetime ends before the score GEMM

  // bf16 conversions (independent of GNN chain)
  k_conv_emb<<<(Vv * HID + 255) / 256, 256, 0, stream>>>(allEmb, embbf, Vv * HID);
  k_conv_wt <<<(MID * HID + 255) / 256, 256, 0, stream>>>(W_fc, Wt);

  // GNN encoder
  k_zero_cnt <<<(Bq * Nn + 255) / 256, 256, 0, stream>>>(cnt, Bq * Nn);
  k_build_adj<<<(Bq * Ee + 255) / 256, 256, 0, stream>>>(esrc, edst, adj, cnt);
  k_mlp_init <<<(Bq * Nn * HID + 255) / 256, 256, 0, stream>>>(h, W_init, b_init, hA);
  k_gin_mean <<<Bq * Nn, 128, 0, stream>>>(hA, hB, adj, cnt);
  k_bn_nodes_relu<<<(Bq * HID + 255) / 256, 256, 0, stream>>>(hB, gamma1, beta1);
  k_gin_mean <<<Bq * Nn, 128, 0, stream>>>(hB, hA, adj, cnt);
  k_bn_nodes_relu<<<(Bq * HID + 255) / 256, 256, 0, stream>>>(hA, gamma2, beta2);
  k_qemb<<<(Bq * HID + 255) / 256, 256, 0, stream>>>(hA, qemb);
  k_qw  <<<(Bq * MID + 255) / 256, 256, 0, stream>>>(qemb, W_fc, b_fc, qw);

  // dominant GEMM on the bf16 WMMA path: 2048 waves, 256 blocks x 8 waves
  k_gemm_score<<<(Bq * Gg) / (8 * 16), 256, 0, stream>>>(embbf, Wt, gPos, qw, Hpre);

  // BN over G + sigmoid head
  k_bn_g_stats<<<(Bq * MID + 255) / 256, 256, 0, stream>>>(Hpre, mu, inv);
  k_head<<<Bq * Gg, MID, 0, stream>>>(Hpre, mu, inv, gamma_bn, beta_bn, W_fc2, b_fc2, probs);
}
